// TriAffineParser_9302899163645
// MI455X (gfx1250) — matrix operand order repaired and checked by reading, not compiled, not run on hardware
//
#include <hip/hip_runtime.h>

// ---------------------------------------------------------------------------
// TriAffine parser for MI455X (gfx1250, wave32, bf16 WMMA 16x16x32).
// B=1, S=128, H=768, D=384, C=8.  Dominant cost: streaming W0/W1 (3.64 GB)
// through HBM exactly once at 23.3 TB/s; all matrix math on v_wmma bf16.
// ---------------------------------------------------------------------------

static constexpr int SEQ  = 128;   // sequence length (x, y, z)
static constexpr int DIM  = 384;   // attention dim (k)
static constexpr int DIMP = 416;   // D+1 padded to 13*32 (i / j contraction dims)
static constexpr int HID  = 768;   // hidden dim
static constexpr int NC   = 8;     // classes

typedef __attribute__((ext_vector_type(8)))  __bf16 bhalf8;
typedef __attribute__((ext_vector_type(16))) __bf16 bhalf16;
typedef __attribute__((ext_vector_type(8)))  float  floatx8;

// ------------------------- WMMA helpers ------------------------------------

static __device__ __forceinline__ floatx8 wmma_bf16(bhalf16 a, bhalf16 b, floatx8 c) {
  // D = A(16x32) * B(32x16) + C(16x16), fp32 accumulate
  return __builtin_amdgcn_wmma_f32_16x16x32_bf16(
      /*neg_a=*/false, a, /*neg_b=*/false, b,
      /*c_mod=*/(short)0, c, /*reuse_a=*/false, /*reuse_b=*/false);
}

// A-matrix fragment (16x32 bf16, row-major source, K contiguous).
// ISA layout: lanes 0-15 hold K = [0..7],[16..23]; lanes 16-31 K = [8..15],[24..31].
static __device__ __forceinline__ bhalf16 frag_a(const __bf16* base, int pitch,
                                                 int col0, int lane) {
  int half = (lane >> 4) & 1, row = lane & 15;
  const __bf16* p = base + (size_t)row * pitch + col0 + 8 * half;
  bhalf8 lo = *(const bhalf8*)(p);
  bhalf8 hi = *(const bhalf8*)(p + 16);
  return __builtin_shufflevector(lo, hi, 0, 1, 2, 3, 4, 5, 6, 7,
                                 8, 9, 10, 11, 12, 13, 14, 15);
}

// B-matrix fragment (32x16 bf16) from B^T storage (N rows, K contiguous).
// ISA layout: lanes 0-15 hold K = [0..7],[16..23]; lanes 16-31 K = [8..15],[24..31].
static __device__ __forceinline__ bhalf16 frag_b(const __bf16* base, int pitch,
                                                 int col0, int lane) {
  int half = (lane >> 4) & 1, row = lane & 15;
  const __bf16* p = base + (size_t)row * pitch + col0 + 8 * half;
  bhalf8 lo = *(const bhalf8*)(p);
  bhalf8 hi = *(const bhalf8*)(p + 16);
  return __builtin_shufflevector(lo, hi, 0, 1, 2, 3, 4, 5, 6, 7,
                                 8, 9, 10, 11, 12, 13, 14, 15);
}

// ------------------------- small prep kernels -------------------------------

__global__ void k_cvt(const float* __restrict__ in, __bf16* __restrict__ out, int n) {
  int i = blockIdx.x * blockDim.x + threadIdx.x;
  if (i < n) out[i] = (__bf16)in[i];
}

// Fill padded columns 384..415 of X and Y (bias col 384 = 1, rest 0).
__global__ void k_pad(__bf16* __restrict__ Xp, __bf16* __restrict__ Yp) {
  int i = blockIdx.x * blockDim.x + threadIdx.x;
  int total = 2 * 2 * SEQ * 32;
  if (i >= total) return;
  int buf = i / (2 * SEQ * 32);
  int rem = i % (2 * SEQ * 32);
  int p   = rem / (SEQ * 32); rem %= SEQ * 32;
  int row = rem / 32;
  int cc  = rem % 32;
  __bf16 v = (__bf16)((cc == 0) ? 1.0f : 0.0f);
  (buf ? Yp : Xp)[((size_t)p * SEQ + row) * DIMP + 384 + cc] = v;
}

// ------------------------- MLP GEMMs (wmma) ---------------------------------
// h[n,s,d] = relu( sum_h mem[s,h] * w1[n,d,h] + b1[n,d] )
__global__ __launch_bounds__(32)
void k_mlp1(const __bf16* __restrict__ memB, const __bf16* __restrict__ w1B,
            const float* __restrict__ b1, __bf16* __restrict__ hB) {
  int nt = blockIdx.x, mt = blockIdx.y, nl = blockIdx.z;
  int lane = threadIdx.x & 31, half = lane >> 4, rowl = lane & 15;
  const __bf16* A  = memB + (size_t)(mt * 16) * HID;
  const __bf16* Bm = w1B + ((size_t)nl * DIM + nt * 16) * HID;   // [d][h] = B^T
  floatx8 c = {};
  for (int ch = 0; ch < HID / 32; ++ch)
    c = wmma_bf16(frag_a(A, HID, ch * 32, lane), frag_b(Bm, HID, ch * 32, lane), c);
#pragma unroll
  for (int g = 0; g < 8; ++g) {
    int s = mt * 16 + g + 8 * half;
    int d = nt * 16 + rowl;
    float v = c[g] + b1[nl * DIM + d];
    hB[((size_t)nl * SEQ + s) * DIM + d] = (__bf16)(v > 0.f ? v : 0.f);
  }
}

// feats[n,s,e] = sum_d h[n,s,d] * w2[n,e,d] + b2[n,e]; routed into padded bufs.
__global__ __launch_bounds__(32)
void k_mlp2(const __bf16* __restrict__ hB, const __bf16* __restrict__ w2B,
            const float* __restrict__ b2, __bf16* __restrict__ Xp,
            __bf16* __restrict__ Yp, __bf16* __restrict__ Zp) {
  int nt = blockIdx.x, mt = blockIdx.y, nl = blockIdx.z;
  int lane = threadIdx.x & 31, half = lane >> 4, rowl = lane & 15;
  const __bf16* A  = hB + (size_t)nl * SEQ * DIM + (size_t)(mt * 16) * DIM;
  const __bf16* Bm = w2B + ((size_t)nl * DIM + nt * 16) * DIM;   // [e][d] = B^T
  floatx8 c = {};
  for (int ch = 0; ch < DIM / 32; ++ch)
    c = wmma_bf16(frag_a(A, DIM, ch * 32, lane), frag_b(Bm, DIM, ch * 32, lane), c);
  int p = nl / 3, which = nl % 3;
  __bf16* dst; int pitch;
  if (which == 0)      { dst = Xp + (size_t)p * SEQ * DIMP; pitch = DIMP; }
  else if (which == 1) { dst = Yp + (size_t)p * SEQ * DIMP; pitch = DIMP; }
  else                 { dst = Zp + (size_t)p * SEQ * DIM;  pitch = DIM;  }
#pragma unroll
  for (int g = 0; g < 8; ++g) {
    int s = mt * 16 + g + 8 * half;
    int e = nt * 16 + rowl;
    dst[(size_t)s * pitch + e] = (__bf16)(c[g] + b2[nl * DIM + e]);
  }
}

// ------------------------- Pass 1: fused X*W_k -> *Y^T ----------------------
// Per workgroup (p, k, r):  U[x,j] = sum_i X[x,i] W[i,k,j,r]   (LDS, bf16)
//                           xwy[x,y] = sum_j U[x,j] Y[y,j]     -> workspace
// r is the fastest grid dim so the 8 sibling WGs share W cachelines (r is the
// fastest memory axis of W): each W element crosses HBM exactly once.
// ldsW is double-buffered (stage chunk c+1 while WMMA consumes chunk c, one
// barrier per chunk), X fragments are held in registers across all j-strips,
// and the next chunk's W cachelines are prefetched into L2 a chunk ahead.
__global__ __launch_bounds__(256)
void k_pass1(const float* __restrict__ W0, const float* __restrict__ W1,
             const __bf16* __restrict__ XpAll, const __bf16* __restrict__ YpAll,
             __bf16* __restrict__ xwy) {
  int bid = blockIdx.x;
  int r = bid & 7;
  int k = (bid >> 3) % DIM;
  int p = bid / (8 * DIM);
  const float*  W = p ? W1 : W0;
  const __bf16* X = XpAll + (size_t)p * SEQ * DIMP;
  const __bf16* Y = YpAll + (size_t)p * SEQ * DIMP;

  __shared__ __align__(16) __bf16 ldsU[SEQ * DIMP];   // 106496 B
  __shared__ __align__(16) __bf16 ldsW[2][32 * 32];   // double-buffered B^T tile

  int tid = threadIdx.x;
  int wave = tid >> 5, lane = tid & 31;
  int half = lane >> 4, rowl = lane & 15;

  // stage W[i0+i, k, j0+n, r] -> dst[n][i] as bf16 (guarded 385x385 extent)
  auto stageW = [&](int ch, int j0, __bf16* dst) {
    int i0 = ch * 32;
#pragma unroll
    for (int q = 0; q < 4; ++q) {
      int e = tid + 256 * q;                   // 0..1023
      int n = e & 31, i = e >> 5;
      int gi = i0 + i, gj = j0 + n;
      float v = 0.f;
      if (gi < 385 && gj < 385)
        v = W[(((size_t)gi * DIM + k) * 385 + gj) * NC + r];
      dst[n * 32 + i] = (__bf16)v;
    }
  };
  // L2 prefetch of a whole chunk (covers all 8 sibling r's: the 32j x 8r span
  // for one i is 1KB contiguous; 32 i's x 8 cachelines = 256 lines = 1/thread)
  auto prefW = [&](int ch, int j0) {
    int i0 = ch * 32;
    int i = tid >> 3, seg = tid & 7;
    int gi = i0 + i;
    if (gi < 385)
      __builtin_prefetch(&W[(((size_t)gi * DIM + k) * 385 + j0) * NC + seg * 32], 0, 1);
  };

  // Preload this wave's X fragments (rows wave*16..+15, all 13 K(i) chunks):
  // reused by every j-strip below.
  bhalf16 afr[DIMP / 32];
#pragma unroll
  for (int ch = 0; ch < DIMP / 32; ++ch)
    afr[ch] = frag_a(X + (size_t)(wave * 16) * DIMP, DIMP, ch * 32, lane);

  // ---- Phase A: U = X (128x416) * W_k (416x416, guarded to 385x385) -------
  for (int st = 0; st < DIMP / 32; ++st) {       // N(j) strips of 32
    int j0 = st * 32;
    floatx8 c0 = {}, c1 = {};
    stageW(0, j0, ldsW[0]);
    prefW(1, j0);
    __syncthreads();
#pragma unroll
    for (int ch = 0; ch < DIMP / 32; ++ch) {     // K(i) chunks of 32
      int cur = ch & 1;
      if (ch + 1 < DIMP / 32) {
        stageW(ch + 1, j0, ldsW[cur ^ 1]);       // overlap stage with WMMA
        if (ch + 2 < DIMP / 32) prefW(ch + 2, j0);
      }
      bhalf16 b0 = frag_b(&ldsW[cur][0], 32, 0, lane);
      bhalf16 b1 = frag_b(&ldsW[cur][16 * 32], 32, 0, lane);
      c0 = wmma_bf16(afr[ch], b0, c0);
      c1 = wmma_bf16(afr[ch], b1, c1);
      __syncthreads();                           // one barrier per chunk
    }
    // store strip into LDS-resident U (bf16), rows owned by this wave
#pragma unroll
    for (int g = 0; g < 8; ++g) {
      int row = wave * 16 + g + 8 * half;
      ldsU[row * DIMP + j0 + rowl]      = (__bf16)c0[g];
      ldsU[row * DIMP + j0 + 16 + rowl] = (__bf16)c1[g];
    }
  }
  __syncthreads();

  // ---- Phase B: xwy = U (128x416) * Y^T (416x128) -------------------------
  floatx8 d[8];
#pragma unroll
  for (int t = 0; t < 8; ++t) d[t] = (floatx8){};
  for (int ch = 0; ch < DIMP / 32; ++ch) {
    int j0 = ch * 32;
    bhalf16 a = frag_a(ldsU + (size_t)(wave * 16) * DIMP, DIMP, j0, lane);
#pragma unroll
    for (int t = 0; t < 8; ++t) {
      bhalf16 b = frag_b(Y + (size_t)(t * 16) * DIMP, DIMP, j0, lane);
      d[t] = wmma_bf16(a, b, d[t]);
    }
  }
  // xwy layout: [p][r][x][y][k]  (k contiguous for pass-2 A fragments)
  size_t base = ((size_t)p * NC + r) * SEQ * SEQ * DIM;
#pragma unroll
  for (int t = 0; t < 8; ++t) {
#pragma unroll
    for (int g = 0; g < 8; ++g) {
      int x = wave * 16 + g + 8 * half;
      int y = t * 16 + rowl;
      xwy[base + ((size_t)x * SEQ + y) * DIM + k] = (__bf16)d[t][g];
    }
  }
}

// ------------------------- Pass 2: score + softmax + final ------------------
// Per workgroup (r, 16 consecutive (x,y) rows): both parsers' score rows over
// the full z range via WMMA, then masked softmax over z, dot, +uni[r].
// The 16-row A blocks (contiguous 12KB per parser) are staged once into LDS
// and shared by all 8 z-tile waves (8x cut in the dominant read stream).
__global__ __launch_bounds__(256)
void k_pass3(const __bf16* __restrict__ xwy, const __bf16* __restrict__ ZpAll,
             const float* __restrict__ uni, float* __restrict__ out) {
  int r  = blockIdx.x & 7;
  int mt = blockIdx.x >> 3;           // 0..1023, covers rows m = mt*16 .. +15
  int m0 = mt * 16;                   // m = x*128 + y
  int x  = m0 >> 7;
  int y0 = m0 & 127;
  int tid = threadIdx.x;
  int wave = tid >> 5, lane = tid & 31;
  int half = lane >> 4, rowl = lane & 15;

  __shared__ __align__(16) __bf16 ldsA[2][16 * DIM];  // 24 KB staged A blocks
  __shared__ float lds0[16][SEQ];
  __shared__ float lds1[16][SEQ];

  const __bf16* A0 = xwy + ((size_t)(0 * NC + r) * SEQ * SEQ + m0) * DIM;
  const __bf16* A1 = xwy + ((size_t)(1 * NC + r) * SEQ * SEQ + m0) * DIM;
  const __bf16* Z0 = ZpAll;
  const __bf16* Z1 = ZpAll + (size_t)SEQ * DIM;
  int z0 = wave * 16;                 // this wave's z tile

  // cooperative stage of both parsers' 16x384 contiguous row blocks (b128)
  {
    const int NV = 16 * DIM / 8;      // v8bf chunks per parser
#pragma unroll
    for (int e = tid; e < 2 * NV; e += 256) {
      int pp = e / NV, idx = e % NV;
      ((bhalf8*)ldsA[0])[e] = pp ? ((const bhalf8*)A1)[idx]
                                 : ((const bhalf8*)A0)[idx];
    }
  }
  __syncthreads();

  floatx8 c0 = {}, c1 = {};
#pragma unroll
  for (int ch = 0; ch < DIM / 32; ++ch) {
    int k0 = ch * 32;
    bhalf16 a0 = frag_a(ldsA[0], DIM, k0, lane);
    bhalf16 a1 = frag_a(ldsA[1], DIM, k0, lane);
    bhalf16 b0 = frag_b(Z0 + (size_t)z0 * DIM, DIM, k0, lane);
    bhalf16 b1 = frag_b(Z1 + (size_t)z0 * DIM, DIM, k0, lane);
    c0 = wmma_bf16(a0, b0, c0);
    c1 = wmma_bf16(a1, b1, c1);
  }
#pragma unroll
  for (int g = 0; g < 8; ++g) {
    int mr = g + 8 * half;
    lds0[mr][z0 + rowl] = c0[g];
    lds1[mr][z0 + rowl] = c1[g];
  }
  __syncthreads();

  // masked softmax over z (score0) and dot with score1; 16 threads per row.
  int row = tid >> 4;                 // 0..15
  int l16 = tid & 15;
  int yy  = y0 + row;
  float mx = -3.4e38f;
  float sc[8];
#pragma unroll
  for (int q = 0; q < 8; ++q) {
    int z = l16 + 16 * q;
    float v = lds0[row][z];
    bool bad = (z > yy) || (z < x);
    v = bad ? -1.0e6f : v;
    sc[q] = v;
    mx = fmaxf(mx, v);
  }
#pragma unroll
  for (int off = 8; off; off >>= 1) mx = fmaxf(mx, __shfl_xor(mx, off, 16));
  float den = 0.f, num = 0.f;
#pragma unroll
  for (int q = 0; q < 8; ++q) {
    int z = l16 + 16 * q;
    float e = __expf(sc[q] - mx);
    den += e;
    num += e * lds1[row][z];
  }
#pragma unroll
  for (int off = 8; off; off >>= 1) {
    den += __shfl_xor(den, off, 16);
    num += __shfl_xor(num, off, 16);
  }
  if (l16 == 0)
    out[((size_t)x * SEQ + yy) * NC + r] = num / den + uni[r];
}

// ------------------------- host launcher ------------------------------------

extern "C" void kernel_launch(void* const* d_in, const int* in_sizes, int n_in,
                              void* d_out, int out_size, void* d_ws, size_t ws_size,
                              hipStream_t stream) {
  const float* memory = (const float*)d_in[0];   // (1,128,768)
  const float* mlp_w1 = (const float*)d_in[1];   // (6,384,768)
  const float* mlp_b1 = (const float*)d_in[2];   // (6,384)
  const float* mlp_w2 = (const float*)d_in[3];   // (6,384,384)
  const float* mlp_b2 = (const float*)d_in[4];   // (6,384)
  const float* W0     = (const float*)d_in[5];   // (385,384,385,8)
  const float* W1     = (const float*)d_in[6];   // (385,384,385,8)
  const float* uni    = (const float*)d_in[7];   // (8,)
  float* out = (float*)d_out;                    // (1,128,128,8)

  // workspace carve-up (256B aligned)
  char* ws = (char*)d_ws;
  size_t off = 0;
  auto carve = [&](size_t bytes) {
    char* p = ws + off;
    off += (bytes + 255) & ~(size_t)255;
    return p;
  };
  __bf16* memB = (__bf16*)carve((size_t)SEQ * HID * 2);
  __bf16* w1B  = (__bf16*)carve((size_t)6 * DIM * HID * 2);
  __bf16* w2B  = (__bf16*)carve((size_t)6 * DIM * DIM * 2);
  __bf16* hB   = (__bf16*)carve((size_t)6 * SEQ * DIM * 2);
  __bf16* Xp   = (__bf16*)carve((size_t)2 * SEQ * DIMP * 2);
  __bf16* Yp   = (__bf16*)carve((size_t)2 * SEQ * DIMP * 2);
  __bf16* Zp   = (__bf16*)carve((size_t)2 * SEQ * DIM * 2);
  __bf16* xwy  = (__bf16*)carve((size_t)2 * NC * SEQ * SEQ * DIM * 2);  // ~201 MB
  (void)ws_size; (void)n_in; (void)in_sizes; (void)out_size;

  // 1) fp32 -> bf16 conversions
  int n0 = SEQ * HID, n1 = 6 * DIM * HID, n2 = 6 * DIM * DIM;
  k_cvt<<<(n0 + 255) / 256, 256, 0, stream>>>(memory, memB, n0);
  k_cvt<<<(n1 + 255) / 256, 256, 0, stream>>>(mlp_w1, w1B, n1);
  k_cvt<<<(n2 + 255) / 256, 256, 0, stream>>>(mlp_w2, w2B, n2);
  k_pad<<<(2 * 2 * SEQ * 32 + 255) / 256, 256, 0, stream>>>(Xp, Yp);

  // 2) MLPs (6 two-layer nets), WMMA
  dim3 g1(DIM / 16, SEQ / 16, 6);
  k_mlp1<<<g1, 32, 0, stream>>>(memB, w1B, mlp_b1, hB);
  k_mlp2<<<g1, 32, 0, stream>>>(hB, w2B, mlp_b2, Xp, Yp, Zp);

  // 3) Pass 1: stream W once, fused (X*W_k)*Y^T -> xwy  (6144 workgroups)
  k_pass1<<<2 * DIM * NC, 256, 0, stream>>>(W0, W1, Xp, Yp, xwy);

  // 4) Pass 2: score via WMMA + masked softmax over z + dot + uni -> out
  k_pass3<<<NC * (SEQ * SEQ / 16), 256, 0, stream>>>(xwy, Zp, uni, out);
}
